// MultiQueryAttention_19765439496782
// MI455X (gfx1250) — matrix operand order repaired and checked by reading, not compiled, not run on hardware
//
#include <hip/hip_runtime.h>
#include <hip/hip_bf16.h>
#include <math.h>
#include <stdint.h>

// ---------------------------------------------------------------------------
// MultiQueryAttention forward for MI455X (gfx1250, wave32, WMMA bf16)
// B=2, T=2048, HID=2048, H=16, D=128  (1 shared KV head)
// ---------------------------------------------------------------------------

#define Bc 2
#define Tc 2048
#define HIDc 2048
#define Hc 16
#define Dc 128

typedef __bf16 bf16_t;
typedef __attribute__((ext_vector_type(16))) __bf16 v16bf;
typedef __attribute__((ext_vector_type(8)))  float  v8f;
typedef int v4i_gcc __attribute__((vector_size(16)));  // matches builtin pointee

#define DEV static __device__ __forceinline__

#if defined(__AMDGCN__) && __has_builtin(__builtin_amdgcn_global_load_async_to_lds_b128)
#define USE_ASYNC_LDS 1
#else
#define USE_ASYNC_LDS 0
#endif

DEV v8f vzero8() {
    v8f z;
#pragma unroll
    for (int i = 0; i < 8; ++i) z[i] = 0.0f;
    return z;
}

DEV float bf2f(bf16_t b) { return (float)b; }
DEV bf16_t f2bf(float f) { return (bf16_t)f; }

// Copy 16 bytes global -> LDS.  Async (ASYNCcnt-tracked) when available.
DEV void copy16_to_lds(const bf16_t* g, bf16_t* l) {
#if USE_ASYNC_LDS
    __builtin_amdgcn_global_load_async_to_lds_b128(
        (__attribute__((address_space(1))) v4i_gcc*)(uintptr_t)g,
        (__attribute__((address_space(3))) v4i_gcc*)(uint32_t)(uintptr_t)l,
        /*offset=*/0, /*cpol=*/0);
#else
    *reinterpret_cast<uint4*>(l) = *reinterpret_cast<const uint4*>(g);
#endif
}

DEV void wait_async0() {
#if USE_ASYNC_LDS
#if __has_builtin(__builtin_amdgcn_s_wait_asynccnt)
    __builtin_amdgcn_s_wait_asynccnt(0);
#else
    asm volatile("s_wait_asynccnt 0" ::: "memory");
#endif
#endif
}

// A-operand fragment (16x32, M x K), row-major source, ld in elements.
// ISA layout: lane m (0-15): K=0..7 (elems 0..7), K=16..23 (elems 8..15)
//             lane m+16   : K=8..15,             K=24..31
DEV v16bf load_afrag(const bf16_t* base, int ld, int lane) {
    const int row = lane & 15;
    const int k0  = (lane >> 4) << 3;
    const bf16_t* p = base + row * ld + k0;
    union { uint4 u[2]; v16bf v; } t;
    t.u[0] = *reinterpret_cast<const uint4*>(p);
    t.u[1] = *reinterpret_cast<const uint4*>(p + 16);
    return t.v;
}

// B-operand fragment (32x16, K x N) given B^T row-major (i.e. column n is
// contiguous along K).  ISA layout: lane n (0-15): K=0..7, K=16..23; lane n+16: K=8..15, K=24..31
DEV v16bf load_bfrag(const bf16_t* base, int ld, int lane) {
    const int col = lane & 15;
    const int k0  = (lane >> 4) << 3;
    const bf16_t* p = base + col * ld + k0;
    union { uint4 u[2]; v16bf v; } t;
    t.u[0] = *reinterpret_cast<const uint4*>(p);
    t.u[1] = *reinterpret_cast<const uint4*>(p + 16);
    return t.v;
}

DEV v8f wmma_bf16(v16bf a, v16bf b, v8f c) {
    return __builtin_amdgcn_wmma_f32_16x16x32_bf16(
        /*neg_a=*/false, a, /*neg_b=*/false, b,
        /*c_mod=*/(short)0, c, /*reuse_a=*/false, /*reuse_b=*/false);
}

DEV float hmax16(float v) {
    v = fmaxf(v, __shfl_xor(v, 1));
    v = fmaxf(v, __shfl_xor(v, 2));
    v = fmaxf(v, __shfl_xor(v, 4));
    v = fmaxf(v, __shfl_xor(v, 8));
    return v;
}
DEV float hsum16(float v) {
    v += __shfl_xor(v, 1);
    v += __shfl_xor(v, 2);
    v += __shfl_xor(v, 4);
    v += __shfl_xor(v, 8);
    return v;
}

// ---------------------------------------------------------------------------
// f32 -> bf16 cast (vectorized)
// ---------------------------------------------------------------------------
__global__ void cast_f32_bf16_kernel(const float* __restrict__ in,
                                     bf16_t* __restrict__ out, int n) {
    int i = (blockIdx.x * blockDim.x + threadIdx.x) * 4;
    if (i + 3 < n) {
        float4 f = *reinterpret_cast<const float4*>(in + i);
        union { bf16_t b[4]; uint2 u; } t;
        t.b[0] = f2bf(f.x); t.b[1] = f2bf(f.y);
        t.b[2] = f2bf(f.z); t.b[3] = f2bf(f.w);
        *reinterpret_cast<uint2*>(out + i) = t.u;
    } else {
        for (; i < n; ++i) out[i] = f2bf(in[i]);
    }
}

// ---------------------------------------------------------------------------
// C[M,N] = A[M,K] (bf16, row-major) @ W[N,K]^T (bf16, row-major) ; OutT output
// Block: 256 threads = 8 waves, tile 64(M) x 256(N); wave tile 32x64.
// K-slabs of 32 double-buffered in LDS via async global->LDS copies; rows
// padded to 40 elements (80B) to spread the 64 LDS banks.
// ---------------------------------------------------------------------------
template <typename OutT>
__global__ __launch_bounds__(256)
void gemm_bt_wmma(const bf16_t* __restrict__ A, const bf16_t* __restrict__ W,
                  OutT* __restrict__ C, int M, int N, int K) {
    __shared__ __align__(16) bf16_t Abuf[2][64][40];
    __shared__ __align__(16) bf16_t Bbuf[2][256][40];

    const int tid  = threadIdx.x;
    const int lane = tid & 31;
    const int wid  = tid >> 5;
    const int mblk = blockIdx.y * 64;
    const int nblk = blockIdx.x * 256;
    const int mw   = (wid >> 2) * 32;  // wave row offset inside block tile
    const int nw   = (wid & 3) * 64;   // wave col offset inside block tile

    // staging assignment: 16B chunks (8 bf16).  A: 64 rows x 4 chunks = 256.
    const int arow = tid >> 2;
    const int acol = (tid & 3) * 8;

    v8f acc[2][4];
#pragma unroll
    for (int i = 0; i < 2; ++i)
#pragma unroll
        for (int j = 0; j < 4; ++j) acc[i][j] = vzero8();

    // prologue: stage K-slab 0 into buffer 0
    copy16_to_lds(A + (size_t)(mblk + arow) * K + acol, &Abuf[0][arow][acol]);
#pragma unroll
    for (int i = 0; i < 4; ++i) {
        const int id = i * 256 + tid;          // B: 256 rows x 4 chunks
        const int brow = id >> 2, bcol = (id & 3) * 8;
        copy16_to_lds(W + (size_t)(nblk + brow) * K + bcol, &Bbuf[0][brow][bcol]);
    }

    int buf = 0;
    for (int kk = 0; kk < K; kk += 32, buf ^= 1) {
        wait_async0();
        __syncthreads();                       // slab[buf] resident + visible
        if (kk + 32 < K) {                     // prefetch next slab into buf^1
            const int kn = kk + 32;
            copy16_to_lds(A + (size_t)(mblk + arow) * K + kn + acol,
                          &Abuf[buf ^ 1][arow][acol]);
#pragma unroll
            for (int i = 0; i < 4; ++i) {
                const int id = i * 256 + tid;
                const int brow = id >> 2, bcol = (id & 3) * 8;
                copy16_to_lds(W + (size_t)(nblk + brow) * K + kn + bcol,
                              &Bbuf[buf ^ 1][brow][bcol]);
            }
        }
        v16bf a0 = load_afrag(&Abuf[buf][mw][0], 40, lane);
        v16bf a1 = load_afrag(&Abuf[buf][mw + 16][0], 40, lane);
#pragma unroll
        for (int j = 0; j < 4; ++j) {
            v16bf b = load_bfrag(&Bbuf[buf][nw + 16 * j][0], 40, lane);
            acc[0][j] = wmma_bf16(a0, b, acc[0][j]);
            acc[1][j] = wmma_bf16(a1, b, acc[1][j]);
        }
        __syncthreads();                       // reads done before buf is rewritten
    }

    const int rb = (lane >> 4) * 8;  // row offset of reg 0 within 16-tile
    const int cl = lane & 15;        // column within 16-tile
#pragma unroll
    for (int i = 0; i < 2; ++i)
#pragma unroll
        for (int j = 0; j < 4; ++j)
#pragma unroll
            for (int r = 0; r < 8; ++r) {
                size_t row = (size_t)(mblk + mw + 16 * i + rb + r);
                C[row * N + (nblk + nw + 16 * j + cl)] = (OutT)acc[i][j][r];
            }
}

// ---------------------------------------------------------------------------
// RoPE for Q.  Reference quirk: apply_rope sees q as (B,H,T,D) but binds its
// "t" axis to H, so the rotation angle is head_index * theta_j (t-independent).
// In:  qraw (B,T,HID) bf16  (col = h*128 + d')
// Out: q    (B,H,T,D) bf16
// ---------------------------------------------------------------------------
__global__ void rope_q_kernel(const bf16_t* __restrict__ qraw,
                              bf16_t* __restrict__ q) {
    int idx = blockIdx.x * blockDim.x + threadIdx.x;  // ((b*H+h)*T+t)*D+d
    if (idx >= Bc * Hc * Tc * Dc) return;
    const int d = idx & 127;
    const int t = (idx >> 7) & 2047;
    const int h = (idx >> 18) & 15;
    const int b = idx >> 22;

    const int half = (d >> 6) & 1;
    const int jj   = d & 63;
    const int j    = d & 31;

    const size_t src = ((size_t)(b * Tc + t)) * HIDc + h * Dc + half * 64 + 2 * j;
    const float xe = bf2f(qraw[src]);
    const float xo = bf2f(qraw[src + 1]);

    // theta_j = 10000^(-2j/64)
    const float theta = __expf(-(float)(2 * j) * (9.210340371976184f / 64.0f));
    const float ang = (float)h * theta;
    const float c = cosf(ang), s = sinf(ang);

    const float out = (jj < 32) ? (xe * c - xo * s) : (xe * s + xo * c);
    q[idx] = f2bf(out);
}

// ---------------------------------------------------------------------------
// RoPE for K (reference passes k as (B,1,T,D): pos axis length 1 -> angle 0,
// i.e. a pure even/odd deinterleave within each 64-wide half) + V transpose.
// In:  kvraw (B,T,256) bf16  (cols 0..127 = k, 128..255 = v)
// Out: k  (B,T,D) bf16 ; vT (B,D,T) bf16
// ---------------------------------------------------------------------------
__global__ void rope_kv_kernel(const bf16_t* __restrict__ kvraw,
                               bf16_t* __restrict__ k,
                               bf16_t* __restrict__ vT) {
    const int t = blockIdx.x & (Tc - 1);
    const int b = blockIdx.x >> 11;
    const int d = threadIdx.x;  // 0..127
    const bf16_t* row = kvraw + (size_t)(b * Tc + t) * 256;

    const int half = d >> 6;
    const int jj   = d & 63;
    const int j    = d & 31;
    const int src  = half * 64 + 2 * j + ((jj >= 32) ? 1 : 0);

    k[(size_t)(b * Tc + t) * Dc + d]    = row[src];
    vT[((size_t)(b * Dc + d)) * Tc + t] = row[128 + d];
}

// ---------------------------------------------------------------------------
// Causal flash attention (multi-query: 1 KV head).
// 1 wave = 16 q-rows x D=128; KV tiles of 32; online softmax in f32.
// q  : (B,H,T,D) bf16 ; k : (B,T,D) bf16 ; vT : (B,D,T) bf16
// out: attn (B,T,HID) bf16   (col = h*128 + d)
// ---------------------------------------------------------------------------
__global__ __launch_bounds__(128)
void flash_mqa_kernel(const bf16_t* __restrict__ q, const bf16_t* __restrict__ k,
                      const bf16_t* __restrict__ vT, bf16_t* __restrict__ attn) {
    __shared__ __align__(16) bf16_t pshare[4][16][32];

    const int lane = threadIdx.x & 31;
    const int wid  = threadIdx.x >> 5;
    const int tile = blockIdx.x * 4 + wid;       // 0 .. B*H*(T/16)-1
    const int qt = tile & 127;
    const int h  = (tile >> 7) & 15;
    const int b  = tile >> 11;
    const int qbase = qt * 16;

    const bf16_t* qp = q  + (((size_t)(b * Hc + h)) * Tc + qbase) * Dc;
    const bf16_t* kp = k  + (size_t)b * Tc * Dc;
    const bf16_t* vp = vT + (size_t)b * Dc * Tc;

    v16bf aq[4];
#pragma unroll
    for (int c = 0; c < 4; ++c) aq[c] = load_afrag(qp + c * 32, Dc, lane);

    float mrow[8], lrow[8];
    v8f o[8];
#pragma unroll
    for (int r = 0; r < 8; ++r) { mrow[r] = -1e30f; lrow[r] = 0.0f; }
#pragma unroll
    for (int oc = 0; oc < 8; ++oc) o[oc] = vzero8();

    const float scaling = 0.08838834764831845f;  // D^-0.5
    const int rb = (lane >> 4) * 8;
    const int cl = lane & 15;
    const int row0 = qbase + rb;
    const int kend = qbase + 16;

    for (int kt = 0; kt < kend; kt += 32) {
        // ---- S = Q @ K^T : 16x32 tile = two 16x16 C-frags over 4 K-steps
        v8f s0 = vzero8(), s1 = vzero8();
#pragma unroll
        for (int c = 0; c < 4; ++c) {
            v16bf bk0 = load_bfrag(kp + (size_t)kt * Dc + c * 32, Dc, lane);
            v16bf bk1 = load_bfrag(kp + (size_t)(kt + 16) * Dc + c * 32, Dc, lane);
            s0 = wmma_bf16(aq[c], bk0, s0);
            s1 = wmma_bf16(aq[c], bk1, s1);
        }

        // ---- scale, causal mask, online softmax (row stats per half-wave)
        const int c0 = kt + cl, c1 = c0 + 16;
        float p0[8], p1[8], scl[8];
#pragma unroll
        for (int r = 0; r < 8; ++r) {
            float x0 = s0[r] * scaling;
            float x1 = s1[r] * scaling;
            if (c0 > row0 + r) x0 = -1e30f;
            if (c1 > row0 + r) x1 = -1e30f;
            float tm = hmax16(fmaxf(x0, x1));
            float mnew = fmaxf(mrow[r], tm);
            float sc = __expf(mrow[r] - mnew);
            float e0 = __expf(x0 - mnew);
            float e1 = __expf(x1 - mnew);
            float ts = hsum16(e0 + e1);
            lrow[r] = lrow[r] * sc + ts;
            mrow[r] = mnew;
            scl[r] = sc;
            p0[r] = e0; p1[r] = e1;
        }
#pragma unroll
        for (int oc = 0; oc < 8; ++oc)
#pragma unroll
            for (int r = 0; r < 8; ++r) o[oc][r] *= scl[r];

        // ---- transpose P (C-layout) -> A-fragment layout via per-wave LDS
#pragma unroll
        for (int r = 0; r < 8; ++r) {
            pshare[wid][rb + r][cl]      = f2bf(p0[r]);
            pshare[wid][rb + r][16 + cl] = f2bf(p1[r]);
        }
        asm volatile("s_wait_dscnt 0" ::: "memory");
        v16bf ap = load_afrag(&pshare[wid][0][0], 32, lane);

        // ---- O += P @ V : 8 output chunks of 16 columns
#pragma unroll
        for (int oc = 0; oc < 8; ++oc) {
            v16bf bv = load_bfrag(vp + (size_t)(oc * 16) * Tc + kt, Tc, lane);
            o[oc] = wmma_bf16(ap, bv, o[oc]);
        }
        asm volatile("s_wait_dscnt 0" ::: "memory");  // reads done before next writes
    }

    // ---- normalize and store to attn (B,T,HID), col = h*128 + d
    bf16_t* op = attn + ((size_t)(b * Tc) + qbase) * HIDc + h * Dc;
#pragma unroll
    for (int oc = 0; oc < 8; ++oc)
#pragma unroll
        for (int r = 0; r < 8; ++r) {
            float val = o[oc][r] / lrow[r];
            op[(size_t)(rb + r) * HIDc + oc * 16 + cl] = f2bf(val);
        }
}

// ---------------------------------------------------------------------------
// Launcher
// ---------------------------------------------------------------------------
extern "C" void kernel_launch(void* const* d_in, const int* in_sizes, int n_in,
                              void* d_out, int out_size, void* d_ws, size_t ws_size,
                              hipStream_t stream) {
    (void)in_sizes; (void)n_in; (void)out_size; (void)ws_size;
    const float* hs = (const float*)d_in[0];
    const float* Wq = (const float*)d_in[1];
    const float* Wk = (const float*)d_in[2];
    const float* Wv = (const float*)d_in[3];
    const float* Wo = (const float*)d_in[4];
    float* out = (float*)d_out;

    char* ws = (char*)d_ws;
    size_t off = 0;
    auto alloc = [&](size_t elems) -> bf16_t* {
        bf16_t* p = (bf16_t*)(ws + off);
        off += ((elems * sizeof(bf16_t)) + 255) & ~(size_t)255;
        return p;
    };

    const size_t nHS  = (size_t)Bc * Tc * HIDc;          // 8,388,608
    const size_t nWq  = (size_t)HIDc * HIDc;             // 4,194,304
    const size_t nWkv = (size_t)256 * HIDc;              //   524,288
    const size_t nKV  = (size_t)Bc * Tc * 256;           // 1,048,576
    const size_t nK   = (size_t)Bc * Tc * Dc;            //   524,288

    bf16_t* hs_bf  = alloc(nHS);
    bf16_t* wq_bf  = alloc(nWq);
    bf16_t* wkv_bf = alloc(nWkv);
    bf16_t* wo_bf  = alloc(nWq);
    bf16_t* qraw   = alloc(nHS);
    bf16_t* kvraw  = alloc(nKV);
    bf16_t* q_bf   = alloc(nHS);
    bf16_t* k_bf   = alloc(nK);
    bf16_t* vT_bf  = alloc(nK);
    bf16_t* attn   = alloc(nHS);

    // 1) casts to bf16 (Wk|Wv fused into one 256x2048 weight)
    cast_f32_bf16_kernel<<<(int)(nHS / 4 / 256), 256, 0, stream>>>(hs, hs_bf, (int)nHS);
    cast_f32_bf16_kernel<<<(int)(nWq / 4 / 256), 256, 0, stream>>>(Wq, wq_bf, (int)nWq);
    cast_f32_bf16_kernel<<<(int)(nWkv / 2 / 4 / 256), 256, 0, stream>>>(Wk, wkv_bf, (int)(nWkv / 2));
    cast_f32_bf16_kernel<<<(int)(nWkv / 2 / 4 / 256), 256, 0, stream>>>(Wv, wkv_bf + nWkv / 2, (int)(nWkv / 2));
    cast_f32_bf16_kernel<<<(int)(nWq / 4 / 256), 256, 0, stream>>>(Wo, wo_bf, (int)nWq);

    const int M = Bc * Tc;  // 4096
    // 2) q projection: qraw = hs @ Wq^T   (4096 x 2048 x 2048)
    gemm_bt_wmma<bf16_t><<<dim3(HIDc / 256, M / 64), 256, 0, stream>>>(
        hs_bf, wq_bf, qraw, M, HIDc, HIDc);
    // 3) k|v projection: kvraw = hs @ [Wk;Wv]^T  (4096 x 256 x 2048)
    gemm_bt_wmma<bf16_t><<<dim3(1, M / 64), 256, 0, stream>>>(
        hs_bf, wkv_bf, kvraw, M, 256, HIDc);
    // 4) RoPE q (head-index angle) -> (B,H,T,D)
    rope_q_kernel<<<(int)(nHS / 256), 256, 0, stream>>>(qraw, q_bf);
    // 5) RoPE k (deinterleave) + V transpose
    rope_kv_kernel<<<Bc * Tc, Dc, 0, stream>>>(kvraw, k_bf, vT_bf);
    // 6) causal flash attention (B*H*T/16 = 4096 wave-tiles, 4 waves/block)
    flash_mqa_kernel<<<(Bc * Hc * (Tc / 16)) / 4, 128, 0, stream>>>(q_bf, k_bf, vT_bf, attn);
    // 7) output projection: out = attn @ Wo^T  (f32 result)
    gemm_bt_wmma<float><<<dim3(HIDc / 256, M / 64), 256, 0, stream>>>(
        attn, wo_bf, out, M, HIDc, HIDc);
}
